// RGAT_56461640073733
// MI455X (gfx1250) — compile-verified, matrix operand order checked by reading, not run on hardware
//
#include <hip/hip_runtime.h>
#include <hip/hip_bf16.h>
#include <math.h>

// ---------------------------------------------------------------------------
// Types
// ---------------------------------------------------------------------------
typedef __bf16 bf16_t;
typedef float  v8f   __attribute__((ext_vector_type(8)));
typedef bf16_t v8bf  __attribute__((ext_vector_type(8)));
typedef bf16_t v16bf __attribute__((ext_vector_type(16)));

union U16 { v16bf v; v8bf h[2]; };

// ---------------------------------------------------------------------------
// Problem constants (fixed by the reference)
// ---------------------------------------------------------------------------
#define NN    20000
#define EE    160000
#define HID_  512
#define HH    8
#define DKK   64
#define LL    4
#define FFD   2048
#define SCALE_ 0.125f   // 1/sqrt(64)
#define LNEPS 1e-5f

// GEMM tiling
#define BM 128
#define BN 128
#define BK 32
#define LDK 40          // padded LDS row (elements) -> conflict-free b128 loads

// bytes per LDS row / buffer segment
#define ROWB   (LDK * 2)                 // 80 bytes
#define ASEG   (BM * LDK)                // elements per A tile
#define BUFSEG ((BM + BN) * LDK)         // elements per double-buffer slot

// ---------------------------------------------------------------------------
// CDNA5 async copy helpers
// ---------------------------------------------------------------------------
__device__ __forceinline__ void async_ld_b128(unsigned lds_off, const void* g)
{
    asm volatile("global_load_async_to_lds_b128 %0, %1, off"
                 :: "v"(lds_off), "v"((unsigned long long)(uintptr_t)g)
                 : "memory");
}

__device__ __forceinline__ void wait_async0()
{
#if __has_builtin(__builtin_amdgcn_s_wait_asynccnt)
    __builtin_amdgcn_s_wait_asynccnt(0);
#else
    asm volatile("s_wait_asynccnt 0x0" ::: "memory");
#endif
}

// ---------------------------------------------------------------------------
// Weight convert + transpose: W (L,K,Nout) f32 -> Wt (L,Nout,K) bf16
// ---------------------------------------------------------------------------
__global__ __launch_bounds__(256)
void transpose_w_kernel(const float* __restrict__ W, bf16_t* __restrict__ Wt,
                        int K, int Nout, int total)
{
    int i = blockIdx.x * 256 + threadIdx.x;
    if (i >= total) return;
    int per = K * Nout;
    int l   = i / per;
    int rem = i - l * per;
    int k   = rem / Nout;
    int n   = rem - k * Nout;
    Wt[(size_t)l * per + (size_t)n * K + k] = (bf16_t)W[i];
}

// x -> f32 working copy + bf16 GEMM input
__global__ __launch_bounds__(256)
void convert_x_kernel(const float* __restrict__ x, float* __restrict__ xf,
                      bf16_t* __restrict__ xb, int total)
{
    int i = blockIdx.x * 256 + threadIdx.x;
    if (i >= total) return;
    float v = x[i];
    xf[i] = v;
    xb[i] = (bf16_t)v;
}

// ---------------------------------------------------------------------------
// WMMA bf16 GEMM:  C[M,Nout] = A[M,K] @ Bt[Nout,K]^T  (+bias, opt relu)
// Double-buffered LDS tiles filled via GLOBAL_LOAD_ASYNC_TO_LDS_B128
// (ASYNCcnt), one barrier per K-step, 8 WMMAs per wave per K-step.
// ---------------------------------------------------------------------------
__global__ __launch_bounds__(256)
void gemm_bf16_wmma(const bf16_t* __restrict__ A,
                    const bf16_t* __restrict__ Bt,
                    const float*  __restrict__ bias,
                    float*        __restrict__ Cf,
                    bf16_t*       __restrict__ Cb,
                    int M, int K, int Nout, int relu)
{
    __shared__ __align__(16) bf16_t smem[2 * BUFSEG];   // 40 KB

    const int tid  = threadIdx.x;
    const int wave = tid >> 5;
    const int lane = tid & 31;
    const int hf   = lane >> 4;   // half of wave: selects K-subblock
    const int l16  = lane & 15;   // M/N index within 16

    const int n0 = blockIdx.x * BN;
    const int m0 = blockIdx.y * BM;

    const int wm = wave >> 1;     // 0..3 : 32-row slab
    const int wn = wave & 1;      // 0..1 : 64-col slab

    // wave-relative LDS byte address of smem (ptrtoint of AS(3) pointer)
    const unsigned ldsBase = (unsigned)(uintptr_t)smem;

    // per-thread staging coordinates: 4 chunks of 8 bf16 per 32-wide row
    const int cr = tid >> 2;            // row 0..63 (+64 on second pass)
    const int cc = (tid & 3) * 8;       // element offset within row

    v8f acc[2][4];
#pragma unroll
    for (int a = 0; a < 2; ++a)
#pragma unroll
        for (int b = 0; b < 4; ++b)
            acc[a][b] = (v8f)0.0f;

    const int nIter = K / BK;

    // issue async copies for tile `it` into buffer `buf`
    auto issue_tile = [&](int buf, int k0) {
        const unsigned aBase = ldsBase + (unsigned)buf * (BUFSEG * 2);
        const unsigned bBase = aBase + (unsigned)(ASEG * 2);
#pragma unroll
        for (int i = 0; i < 2; ++i) {
            int r  = cr + i * 64;
            int ga = m0 + r; if (ga >= M) ga = M - 1;   // clamp tail (rows never stored)
            async_ld_b128(aBase + (unsigned)r * ROWB + (unsigned)cc * 2,
                          A + (size_t)ga * K + k0 + cc);
            async_ld_b128(bBase + (unsigned)r * ROWB + (unsigned)cc * 2,
                          Bt + (size_t)(n0 + r) * K + k0 + cc);
        }
    };

    issue_tile(0, 0);   // prologue

    for (int it = 0; it < nIter; ++it) {
        const int buf = it & 1;
        wait_async0();        // own async slice landed in LDS
        __syncthreads();      // everyone's slice landed; prev fragment reads drained
        if (it + 1 < nIter) issue_tile(buf ^ 1, (it + 1) * BK);

        const int aOff = buf * BUFSEG;
        const int bOff = aOff + ASEG;

        v16bf af[2], bfr[4];
#pragma unroll
        for (int tm = 0; tm < 2; ++tm) {
            int row = wm * 32 + tm * 16 + l16;
            int e   = aOff + row * LDK + hf * 8;
            U16 u;
            u.h[0] = *(const v8bf*)(&smem[e]);
            u.h[1] = *(const v8bf*)(&smem[e + 16]);
            af[tm] = u.v;
        }
#pragma unroll
        for (int tn = 0; tn < 4; ++tn) {
            int row = wn * 64 + tn * 16 + l16;
            int e   = bOff + row * LDK + hf * 8;
            U16 u;
            u.h[0] = *(const v8bf*)(&smem[e]);
            u.h[1] = *(const v8bf*)(&smem[e + 16]);
            bfr[tn] = u.v;
        }
#pragma unroll
        for (int tm = 0; tm < 2; ++tm)
#pragma unroll
            for (int tn = 0; tn < 4; ++tn)
                acc[tm][tn] = __builtin_amdgcn_wmma_f32_16x16x32_bf16(
                    false, af[tm], false, bfr[tn], (short)0, acc[tm][tn],
                    false, false);
    }

    // ---- epilogue: bias, relu, f32/bf16 stores ----
#pragma unroll
    for (int tm = 0; tm < 2; ++tm) {
#pragma unroll
        for (int tn = 0; tn < 4; ++tn) {
            int ncol = n0 + wn * 64 + tn * 16 + l16;
            float bv = bias ? bias[ncol] : 0.0f;
#pragma unroll
            for (int r = 0; r < 8; ++r) {
                int mrow = m0 + wm * 32 + tm * 16 + r + hf * 8;
                if (mrow < M) {
                    float v = acc[tm][tn][r] + bv;
                    if (relu) v = fmaxf(v, 0.0f);
                    size_t o = (size_t)mrow * Nout + ncol;
                    if (Cf) Cf[o] = v;
                    if (Cb) Cb[o] = (bf16_t)v;
                }
            }
        }
    }
}

// ---------------------------------------------------------------------------
// Attention scratch init: wv=0, z=0, smax=-inf
// ---------------------------------------------------------------------------
__global__ __launch_bounds__(256)
void init_attn_kernel(float* __restrict__ smax, float* __restrict__ z,
                      float* __restrict__ wv)
{
    int i = blockIdx.x * 256 + threadIdx.x;
    if (i < NN * HID_) wv[i] = 0.0f;
    if (i < NN * HH) { smax[i] = -INFINITY; z[i] = 0.0f; }
}

__device__ __forceinline__ void atomicMaxF32(float* addr, float val)
{
    if (val >= 0.0f) atomicMax((int*)addr, __float_as_int(val));
    else             atomicMin((unsigned int*)addr, __float_as_uint(val));
}

// ---------------------------------------------------------------------------
// Edge scores: one wave per (edge, head); score = ((k[src]+r).q[dst])*scale
// plus segment-max into smax[dst].
// ---------------------------------------------------------------------------
__global__ __launch_bounds__(256)
void edge_score_kernel(const float* __restrict__ q, const float* __restrict__ k,
                       const float* __restrict__ rel,
                       const int* __restrict__ relations,
                       const int* __restrict__ src, const int* __restrict__ dst,
                       float* __restrict__ score, float* __restrict__ smax)
{
    int w = blockIdx.x * 8 + (threadIdx.x >> 5);
    if (w >= EE * HH) return;
    int lane = threadIdx.x & 31;
    int e = w >> 3, h = w & 7;
    int s_ = src[e], d_ = dst[e], r_ = relations[e];
    size_t ko = (size_t)s_ * HID_ + h * DKK;
    size_t qo = (size_t)d_ * HID_ + h * DKK;
    size_t ro = (size_t)r_ * HID_ + h * DKK;
    float acc = (k[ko + lane]      + rel[ro + lane])      * q[qo + lane]
              + (k[ko + 32 + lane] + rel[ro + 32 + lane]) * q[qo + 32 + lane];
#pragma unroll
    for (int off = 16; off; off >>= 1) acc += __shfl_down(acc, off);
    if (lane == 0) {
        float sc = acc * SCALE_;
        score[(size_t)e * HH + h] = sc;
        atomicMaxF32(&smax[(size_t)d_ * HH + h], sc);
    }
}

// ---------------------------------------------------------------------------
// Edge aggregate: ex = exp(score - smax'); z[dst]+=ex; wv[dst]+=ex*(v[src]+r)
// ---------------------------------------------------------------------------
__global__ __launch_bounds__(256)
void edge_agg_kernel(const float* __restrict__ v, const float* __restrict__ rel,
                     const int* __restrict__ relations,
                     const int* __restrict__ src, const int* __restrict__ dst,
                     const float* __restrict__ score,
                     const float* __restrict__ smax,
                     float* __restrict__ z, float* __restrict__ wv)
{
    int w = blockIdx.x * 8 + (threadIdx.x >> 5);
    if (w >= EE * HH) return;
    int lane = threadIdx.x & 31;
    int e = w >> 3, h = w & 7;
    int s_ = src[e], d_ = dst[e], r_ = relations[e];
    float m = smax[(size_t)d_ * HH + h];
    if (m < -3.0e38f) m = 0.0f;                  // nodes w/o in-edges
    float ex = __expf(score[(size_t)e * HH + h] - m);
    if (lane == 0) atomicAdd(&z[(size_t)d_ * HH + h], ex);
    size_t vo = (size_t)s_ * HID_ + h * DKK;
    size_t ro = (size_t)r_ * HID_ + h * DKK;
    size_t wo = (size_t)d_ * HID_ + h * DKK;
    atomicAdd(&wv[wo + lane],      ex * (v[vo + lane]      + rel[ro + lane]));
    atomicAdd(&wv[wo + 32 + lane], ex * (v[vo + 32 + lane] + rel[ro + 32 + lane]));
}

// o = wv / (z + 1e-9)  -> bf16 input for the O-projection GEMM
__global__ __launch_bounds__(256)
void finalize_o_kernel(const float* __restrict__ wv, const float* __restrict__ z,
                       bf16_t* __restrict__ ob)
{
    int i = blockIdx.x * 256 + threadIdx.x;
    if (i >= NN * HID_) return;
    int n = i >> 9;
    int h = (i >> 6) & 7;
    float zz = z[(size_t)n * HH + h];
    ob[i] = (bf16_t)(wv[i] / (zz + 1e-9f));
}

// ---------------------------------------------------------------------------
// out = LayerNorm(A + B) * g + b; one wave per 512-wide row
// ---------------------------------------------------------------------------
__global__ __launch_bounds__(256)
void resid_ln_kernel(const float* __restrict__ A, const float* __restrict__ B,
                     const float* __restrict__ g, const float* __restrict__ bb,
                     float* __restrict__ outF, bf16_t* __restrict__ outB,
                     int Nrows)
{
    int row = blockIdx.x * 8 + (threadIdx.x >> 5);
    if (row >= Nrows) return;
    int lane = threadIdx.x & 31;
    size_t base = (size_t)row * HID_;
    float v[16];
    float s = 0.0f;
#pragma unroll
    for (int j = 0; j < 16; ++j) {
        int c = j * 32 + lane;
        v[j] = A[base + c] + B[base + c];
        s += v[j];
    }
#pragma unroll
    for (int off = 16; off; off >>= 1) s += __shfl_down(s, off);
    s = __shfl(s, 0);
    float mean = s * (1.0f / HID_);
    float vs = 0.0f;
#pragma unroll
    for (int j = 0; j < 16; ++j) { float d = v[j] - mean; vs += d * d; }
#pragma unroll
    for (int off = 16; off; off >>= 1) vs += __shfl_down(vs, off);
    vs = __shfl(vs, 0);
    float rstd = rsqrtf(vs * (1.0f / HID_) + LNEPS);
#pragma unroll
    for (int j = 0; j < 16; ++j) {
        int c = j * 32 + lane;
        float o = (v[j] - mean) * rstd * g[c] + bb[c];
        outF[base + c] = o;
        if (outB) outB[base + c] = (bf16_t)o;
    }
}

// ---------------------------------------------------------------------------
// Host launcher
// ---------------------------------------------------------------------------
extern "C" void kernel_launch(void* const* d_in, const int* in_sizes, int n_in,
                              void* d_out, int out_size, void* d_ws, size_t ws_size,
                              hipStream_t stream)
{
    const float* x         = (const float*)d_in[0];
    const int*   relations = (const int*)  d_in[1];
    const int*   srcI      = (const int*)  d_in[2];
    const int*   dstI      = (const int*)  d_in[3];
    const float* rel_embed = (const float*)d_in[4];
    const float* Wq = (const float*)d_in[5];  const float* bq = (const float*)d_in[6];
    const float* Wk = (const float*)d_in[7];  const float* bk = (const float*)d_in[8];
    const float* Wv = (const float*)d_in[9];  const float* bv = (const float*)d_in[10];
    const float* Wo = (const float*)d_in[11]; const float* bo = (const float*)d_in[12];
    const float* W1 = (const float*)d_in[13]; const float* b1 = (const float*)d_in[14];
    const float* W2 = (const float*)d_in[15]; const float* b2 = (const float*)d_in[16];
    const float* ln1g = (const float*)d_in[17]; const float* ln1b = (const float*)d_in[18];
    const float* ln2g = (const float*)d_in[19]; const float* ln2b = (const float*)d_in[20];
    float* out = (float*)d_out;

    // workspace sub-allocation (deterministic, 256B aligned)
    char* wsp = (char*)d_ws;
    size_t off = 0;
    auto alloc = [&](size_t bytes) -> void* {
        void* p = wsp + off;
        off = (off + bytes + 255) & ~(size_t)255;
        return p;
    };
    bf16_t* wqT  = (bf16_t*)alloc((size_t)LL * HID_ * HID_ * 2);
    bf16_t* wkT  = (bf16_t*)alloc((size_t)LL * HID_ * HID_ * 2);
    bf16_t* wvT  = (bf16_t*)alloc((size_t)LL * HID_ * HID_ * 2);
    bf16_t* woT  = (bf16_t*)alloc((size_t)LL * HID_ * HID_ * 2);
    bf16_t* w1T  = (bf16_t*)alloc((size_t)LL * HID_ * FFD * 2);
    bf16_t* w2T  = (bf16_t*)alloc((size_t)LL * FFD * HID_ * 2);
    float*  xcur = (float*) alloc((size_t)NN * HID_ * 4);
    bf16_t* xb   = (bf16_t*)alloc((size_t)NN * HID_ * 2);
    float*  qb   = (float*) alloc((size_t)NN * HID_ * 4);
    float*  kb   = (float*) alloc((size_t)NN * HID_ * 4);
    float*  vb   = (float*) alloc((size_t)NN * HID_ * 4);
    float*  wvag = (float*) alloc((size_t)NN * HID_ * 4);
    float*  smax = (float*) alloc((size_t)NN * HH * 4);
    float*  zsum = (float*) alloc((size_t)NN * HH * 4);
    float*  score= (float*) alloc((size_t)EE * HH * 4);
    bf16_t* ob   = (bf16_t*)alloc((size_t)NN * HID_ * 2);
    float*  proj = (float*) alloc((size_t)NN * HID_ * 4);  // O-proj out, then FF2 out
    float*  hcur = (float*) alloc((size_t)NN * HID_ * 4);
    bf16_t* hb   = (bf16_t*)alloc((size_t)NN * HID_ * 2);
    bf16_t* ff1b = (bf16_t*)alloc((size_t)NN * FFD * 2);

    const int elemBlocks = (NN * HID_ + 255) / 256;
    const int edgeBlocks = (EE * HH + 7) / 8;
    const int lnBlocks   = (NN + 7) / 8;
    const dim3 g512(HID_ / BN, (NN + BM - 1) / BM);   // GEMM, Nout=512
    const dim3 g2048(FFD / BN, (NN + BM - 1) / BM);   // GEMM, Nout=2048

    // one-time (per launch) weight convert+transpose to bf16
    {
        int t = LL * HID_ * HID_;
        int tb = (t + 255) / 256;
        transpose_w_kernel<<<tb, 256, 0, stream>>>(Wq, wqT, HID_, HID_, t);
        transpose_w_kernel<<<tb, 256, 0, stream>>>(Wk, wkT, HID_, HID_, t);
        transpose_w_kernel<<<tb, 256, 0, stream>>>(Wv, wvT, HID_, HID_, t);
        transpose_w_kernel<<<tb, 256, 0, stream>>>(Wo, woT, HID_, HID_, t);
        int t2 = LL * HID_ * FFD;
        int tb2 = (t2 + 255) / 256;
        transpose_w_kernel<<<tb2, 256, 0, stream>>>(W1, w1T, HID_, FFD, t2);
        transpose_w_kernel<<<tb2, 256, 0, stream>>>(W2, w2T, FFD, HID_, t2);
    }

    convert_x_kernel<<<elemBlocks, 256, 0, stream>>>(x, xcur, xb, NN * HID_);

    for (int l = 0; l < LL; ++l) {
        const size_t ws2 = (size_t)l * HID_ * HID_;
        const size_t wf  = (size_t)l * HID_ * FFD;
        // Q/K/V projections (bf16 WMMA, f32 out)
        gemm_bf16_wmma<<<g512, 256, 0, stream>>>(xb, wqT + ws2, bq + l * HID_,
                                                 qb, nullptr, NN, HID_, HID_, 0);
        gemm_bf16_wmma<<<g512, 256, 0, stream>>>(xb, wkT + ws2, bk + l * HID_,
                                                 kb, nullptr, NN, HID_, HID_, 0);
        gemm_bf16_wmma<<<g512, 256, 0, stream>>>(xb, wvT + ws2, bv + l * HID_,
                                                 vb, nullptr, NN, HID_, HID_, 0);
        // graph attention
        init_attn_kernel<<<elemBlocks, 256, 0, stream>>>(smax, zsum, wvag);
        edge_score_kernel<<<edgeBlocks, 256, 0, stream>>>(qb, kb, rel_embed,
                                                          relations, srcI, dstI,
                                                          score, smax);
        edge_agg_kernel<<<edgeBlocks, 256, 0, stream>>>(vb, rel_embed, relations,
                                                        srcI, dstI, score, smax,
                                                        zsum, wvag);
        finalize_o_kernel<<<elemBlocks, 256, 0, stream>>>(wvag, zsum, ob);
        // O projection
        gemm_bf16_wmma<<<g512, 256, 0, stream>>>(ob, woT + ws2, bo + l * HID_,
                                                 proj, nullptr, NN, HID_, HID_, 0);
        // h = LN1(x + o)
        resid_ln_kernel<<<lnBlocks, 256, 0, stream>>>(xcur, proj,
                                                      ln1g + l * HID_, ln1b + l * HID_,
                                                      hcur, hb, NN);
        // FFN: relu(h@W1+b1)@W2+b2
        gemm_bf16_wmma<<<g2048, 256, 0, stream>>>(hb, w1T + wf, b1 + l * FFD,
                                                  nullptr, ff1b, NN, HID_, FFD, 1);
        gemm_bf16_wmma<<<g512, 256, 0, stream>>>(ff1b, w2T + wf, b2 + l * HID_,
                                                 proj, nullptr, NN, FFD, HID_, 0);
        // x = LN2(h + ff); last layer writes straight to d_out
        bool last = (l == LL - 1);
        resid_ln_kernel<<<lnBlocks, 256, 0, stream>>>(hcur, proj,
                                                      ln2g + l * HID_, ln2b + l * HID_,
                                                      last ? out : xcur,
                                                      last ? nullptr : xb, NN);
    }
}
